// FocusedLinearAttention_29033978921046
// MI455X (gfx1250) — compile-verified, matrix operand order checked by reading, not run on hardware
//
#include <hip/hip_runtime.h>

typedef __attribute__((ext_vector_type(16))) _Float16 v16h;
typedef __attribute__((ext_vector_type(8)))  float    v8f;
typedef __attribute__((ext_vector_type(2)))  float    v2f;

constexpr int B  = 128, N = 1024, C = 128, H = 8, HD = 16;
constexpr int BN = B * N;        // 131072 tokens
constexpr int C3 = 3 * C;        // 384
constexpr float EPS = 1e-6f;
constexpr int TM = 64;           // token tile for the big GEMMs

// ---------------------------------------------------------------------------
// Async global->LDS tile copy (GLOBAL_LOAD_ASYNC_TO_LDS_B128, ASYNCcnt).
// ISA 10.2: generic pointers in the LDS aperture hold the LDS byte offset in
// bits [31:0], so truncation gives the LDS address VGPR for the async op.
// ---------------------------------------------------------------------------
__device__ __forceinline__ void async_copy_tile(const float* __restrict__ gsrc,
                                                float* lds_dst, int bytes,
                                                int tid) {
  const unsigned lbase = (unsigned)(unsigned long long)lds_dst;
  const char* gb = (const char*)gsrc;
  for (int off = tid * 16; off < bytes; off += 256 * 16) {
    const unsigned l = lbase + (unsigned)off;
    const char* ga = gb + off;
    asm volatile("global_load_async_to_lds_b128 %0, %1, off"
                 :: "v"(l), "v"(ga) : "memory");
  }
  asm volatile("s_wait_asynccnt 0x0" ::: "memory");
}

// ---------------------------------------------------------------------------
// Kernel 0: one-shot weight conversion to f16 (removes per-block v_cvt work).
// ---------------------------------------------------------------------------
__global__ __launch_bounds__(256)
void cvt_weights_kernel(const float* __restrict__ wq,
                        const float* __restrict__ wp,
                        _Float16* __restrict__ wq_h,
                        _Float16* __restrict__ wp_h) {
  const int i = blockIdx.x * 256 + threadIdx.x;
  if (i < C3 * C) wq_h[i] = (_Float16)wq[i];
  if (i < C * C)  wp_h[i] = (_Float16)wp[i];
}

// ---------------------------------------------------------------------------
// Kernel 1: fused QKV GEMM (f16 WMMA, f32 accum) + focusing transform.
// 64 tokens x 384 outputs per block; async-staged A tile; epilogue does the
// per-token focusing transform in LDS and writes q/k/v head-major.
// ---------------------------------------------------------------------------
__global__ __launch_bounds__(256)
void qkv_focus_kernel(const float* __restrict__ x,
                      const float* __restrict__ pos,
                      const float* __restrict__ mask,
                      const float* __restrict__ scale_param,
                      const _Float16* __restrict__ wq_h,
                      const float* __restrict__ b_qkv,
                      float* __restrict__ qh,
                      float* __restrict__ kh,
                      float* __restrict__ vh) {
  __shared__ __align__(16) _Float16 As[TM][C + 8];   // x tile, fp16 (~17 KB)
  __shared__ __align__(16) float    Os[TM][C3];      // qkv tile, fp32 (96 KB)
  __shared__ float sp[C];                            // softplus(scale_param)

  const int tid  = threadIdx.x;
  const int lane = tid & 31;
  const int wave = tid >> 5;
  const int t0   = blockIdx.x * TM;

  // Stage x tile (f32) into Os scratch via async LDS loads, then widen->f16.
  async_copy_tile(x + (size_t)t0 * C, &Os[0][0], TM * C * 4, tid);
  if (tid < C) sp[tid] = log1pf(__expf(scale_param[tid]));
  __syncthreads();
  const float* stage = &Os[0][0];
  for (int i = tid; i < TM * C; i += 256)
    As[i >> 7][i & (C - 1)] = (_Float16)stage[i];
  __syncthreads();

  const int kb   = (lane < 16) ? 0 : 8;  // 16-bit A/B K-base per lane half
  const int mcol = lane & 15;
  const int mo   = (lane < 16) ? 0 : 8;  // f32 D row offset per lane half

  // (TM/16) x (C3/16) = 4 x 24 = 96 output tiles over 8 waves
  for (int tile = wave; tile < (TM / 16) * (C3 / 16); tile += 8) {
    const int m0   = (tile % (TM / 16)) * 16;
    const int n0   = (tile / (TM / 16)) * 16;
    const int ncol = n0 + mcol;
    const _Float16* wr = wq_h + (size_t)ncol * C + kb;
    v8f acc = {};
#pragma unroll
    for (int kk = 0; kk < C; kk += 32) {
      v16h a, b;
#pragma unroll
      for (int e = 0; e < 16; ++e) {
        const int k = e + (e >= 8 ? 8 : 0);   // ISA 16-bit fragment K pattern
        a[e] = As[m0 + mcol][kk + kb + k];
        b[e] = wr[kk + k];                    // B[k][n] = w_qkv[n][k], fp16
      }
      acc = __builtin_amdgcn_wmma_f32_16x16x32_f16(false, a, false, b,
                                                   (short)0, acc, false, false);
    }
    const float bias = b_qkv[ncol];
#pragma unroll
    for (int r = 0; r < 8; ++r)
      Os[m0 + r + mo][ncol] = acc[r] + bias;
  }
  __syncthreads();

  // Epilogue: 4 roles x 64 tokens over 256 threads.
  const int t    = tid & (TM - 1);
  const int role = tid / TM;                 // 0:q 1:k 2:v[0:64) 3:v[64:128)
  const int g    = t0 + t;
  const int bb   = g / N;
  const int nn   = g & (N - 1);

  if (role < 2) {
    const int off = role * C;
    const float* po = pos + (size_t)g * C;
    float s2 = 0.f;
    for (int c = 0; c < C; ++c) {
      float v = Os[t][off + c] + po[c];
      v = fmaxf(v, 0.f) + EPS;
      v = v / sp[c];
      Os[t][off + c] = v;
      s2 += v * v;
    }
    const float qn = sqrtf(s2);
    float s6 = 0.f;
    for (int c = 0; c < C; ++c) {
      const float v = Os[t][off + c];
      const float v3 = v * v * v;
      s6 += v3 * v3;
    }
    float scl = qn * rsqrtf(s6);             // q^3 / ||q^3|| * ||q||
    if (role == 1) scl *= mask[g];
    float* dst = (role == 0) ? qh : kh;
    for (int c = 0; c < C; ++c) {
      const float v = Os[t][off + c];
      const int h = c >> 4, d = c & 15;
      dst[((size_t)(bb * H + h) * N + nn) * HD + d] = v * v * v * scl;
    }
  } else {
    const int cb = (role - 2) * 64;
    for (int c = cb; c < cb + 64; ++c) {
      const int h = c >> 4, d = c & 15;
      vh[((size_t)(bb * H + h) * N + nn) * HD + d] = Os[t][2 * C + c];
    }
  }
}

// ---------------------------------------------------------------------------
// Kernel 2: per-(b,h) kv = K^T V (16x16) via f32 WMMA 16x16x4, plus ksum.
// ---------------------------------------------------------------------------
__global__ __launch_bounds__(256)
void kv_ksum_kernel(const float* __restrict__ kh,
                    const float* __restrict__ vh,
                    float* __restrict__ kv_g,
                    float* __restrict__ ksum_g) {
  __shared__ float part[8][256];
  __shared__ float ksum_s[16];
  const int bh   = blockIdx.x;
  const int tid  = threadIdx.x;
  const int lane = tid & 31;
  const int wave = tid >> 5;
  if (tid < 16) ksum_s[tid] = 0.f;
  __syncthreads();

  const int ch = lane & 15;
  const int ko = (lane < 16) ? 0 : 2;        // 32-bit A/B K-base per lane half
  const int mo = (lane < 16) ? 0 : 8;
  const float* kb_ = kh + (size_t)bh * N * HD;
  const float* vb_ = vh + (size_t)bh * N * HD;

  v8f acc = {};
  for (int n = wave * 128; n < wave * 128 + 128; n += 4) {
    v2f a, b;
#pragma unroll
    for (int e = 0; e < 2; ++e) {
      a[e] = kb_[(size_t)(n + ko + e) * HD + ch];   // A = K^T (c x n)
      b[e] = vb_[(size_t)(n + ko + e) * HD + ch];   // B = V   (n x d)
    }
    acc = __builtin_amdgcn_wmma_f32_16x16x4_f32(false, a, false, b,
                                                (short)0, acc, false, false);
  }
#pragma unroll
  for (int r = 0; r < 8; ++r)
    part[wave][(r + mo) * 16 + ch] = acc[r];

  // ksum partials: thread = (64-token chunk, channel)
  float ps = 0.f;
  const int chunk = tid >> 4, cc = tid & 15;
  for (int n = chunk * 64; n < chunk * 64 + 64; ++n)
    ps += kb_[(size_t)n * HD + cc];
  atomicAdd(&ksum_s[cc], ps);
  __syncthreads();

  float s = 0.f;
#pragma unroll
  for (int w = 0; w < 8; ++w) s += part[w][tid];
  kv_g[(size_t)bh * 256 + tid] = s;
  if (tid < 16) ksum_g[bh * 16 + tid] = ksum_s[tid];
}

// ---------------------------------------------------------------------------
// Kernel 3: o = (Q @ kv) * z, scattered via voxel_inds into projection order.
// ---------------------------------------------------------------------------
__global__ __launch_bounds__(256)
void attn_out_kernel(const float* __restrict__ qh,
                     const float* __restrict__ kv_g,
                     const float* __restrict__ ksum_g,
                     const int* __restrict__ vox,
                     float* __restrict__ o_pre) {
  __shared__ float z_s[N];
  __shared__ float kv_s[256];
  __shared__ float ks_s[16];
  const int bh   = blockIdx.x;
  const int tid  = threadIdx.x;
  const int lane = tid & 31;
  const int wave = tid >> 5;
  kv_s[tid] = kv_g[(size_t)bh * 256 + tid];
  if (tid < 16) ks_s[tid] = ksum_g[bh * 16 + tid];
  __syncthreads();

  const float* qb = qh + (size_t)bh * N * HD;
  for (int n = tid; n < N; n += 256) {
    float dot = 0.f;
#pragma unroll
    for (int c = 0; c < HD; ++c) dot += qb[(size_t)n * HD + c] * ks_s[c];
    z_s[n] = 1.f / (dot + EPS);
  }
  __syncthreads();

  const int ch = lane & 15;
  const int ko = (lane < 16) ? 0 : 2;
  const int mo = (lane < 16) ? 0 : 8;

  v2f bf[4];                                  // kv as B fragments (K=16 in 4x4)
#pragma unroll
  for (int kk = 0; kk < 4; ++kk)
#pragma unroll
    for (int e = 0; e < 2; ++e)
      bf[kk][e] = kv_s[(kk * 4 + ko + e) * 16 + ch];

  const int bb = bh >> 3;
  const int h  = bh & 7;
  for (int m0 = wave * 16; m0 < N; m0 += 8 * 16) {
    v8f acc = {};
#pragma unroll
    for (int kk = 0; kk < 4; ++kk) {
      v2f a;
#pragma unroll
      for (int e = 0; e < 2; ++e)
        a[e] = qb[(size_t)(m0 + ch) * HD + kk * 4 + ko + e];
      acc = __builtin_amdgcn_wmma_f32_16x16x4_f32(false, a, false, bf[kk],
                                                  (short)0, acc, false, false);
    }
#pragma unroll
    for (int r = 0; r < 8; ++r) {
      const int row = m0 + r + mo;
      const int p   = vox[bb * N + row];       // scatter: dest row = flat[g]
      o_pre[(size_t)p * C + h * HD + ch] = acc[r] * z_s[row];
    }
  }
}

// ---------------------------------------------------------------------------
// Kernel 4: final projection GEMM (f16 WMMA, f32 accum) + bias -> d_out.
// ---------------------------------------------------------------------------
__global__ __launch_bounds__(256)
void proj_kernel(const float* __restrict__ o_pre,
                 const _Float16* __restrict__ wp_h,
                 const float* __restrict__ b_proj,
                 float* __restrict__ out) {
  __shared__ __align__(16) _Float16 As[TM][C + 8];
  __shared__ __align__(16) float    stage[TM * C];   // 32 KB staging
  const int tid  = threadIdx.x;
  const int lane = tid & 31;
  const int wave = tid >> 5;
  const int t0   = blockIdx.x * TM;

  async_copy_tile(o_pre + (size_t)t0 * C, stage, TM * C * 4, tid);
  __syncthreads();
  for (int i = tid; i < TM * C; i += 256)
    As[i >> 7][i & (C - 1)] = (_Float16)stage[i];
  __syncthreads();

  const int kb   = (lane < 16) ? 0 : 8;
  const int mcol = lane & 15;
  const int mo   = (lane < 16) ? 0 : 8;

  // (TM/16) x (C/16) = 4 x 8 = 32 tiles over 8 waves
  for (int tile = wave; tile < (TM / 16) * (C / 16); tile += 8) {
    const int m0   = (tile % (TM / 16)) * 16;
    const int n0   = (tile / (TM / 16)) * 16;
    const int ncol = n0 + mcol;
    const _Float16* wr = wp_h + (size_t)ncol * C + kb;
    v8f acc = {};
#pragma unroll
    for (int kk = 0; kk < C; kk += 32) {
      v16h a, b;
#pragma unroll
      for (int e = 0; e < 16; ++e) {
        const int k = e + (e >= 8 ? 8 : 0);
        a[e] = As[m0 + mcol][kk + kb + k];
        b[e] = wr[kk + k];
      }
      acc = __builtin_amdgcn_wmma_f32_16x16x32_f16(false, a, false, b,
                                                   (short)0, acc, false, false);
    }
    const float bias = b_proj[ncol];
#pragma unroll
    for (int r = 0; r < 8; ++r)
      out[(size_t)(t0 + m0 + r + mo) * C + ncol] = acc[r] + bias;
  }
}

// ---------------------------------------------------------------------------
extern "C" void kernel_launch(void* const* d_in, const int* in_sizes, int n_in,
                              void* d_out, int out_size, void* d_ws, size_t ws_size,
                              hipStream_t stream) {
  const float* x           = (const float*)d_in[0];
  const float* pos         = (const float*)d_in[1];
  const float* mask        = (const float*)d_in[2];
  const float* scale_param = (const float*)d_in[3];
  const float* w_qkv       = (const float*)d_in[4];
  const float* b_qkv       = (const float*)d_in[5];
  const float* w_proj      = (const float*)d_in[6];
  const float* b_proj      = (const float*)d_in[7];
  const int*   vox         = (const int*)d_in[9];   // voxel_inds (flattened)
  float*       out         = (float*)d_out;

  float* ws = (float*)d_ws;
  const size_t TOK = (size_t)BN * C;                 // 16,777,216 floats
  float* qh    = ws;                                 // (B*H, N, HD)
  float* kh    = qh + TOK;
  float* vh    = kh + TOK;
  float* o_pre = vh + TOK;                           // (B*N, C), permuted rows
  float* kv_g  = o_pre + TOK;                        // (B*H, 16, 16)
  float* ksum  = kv_g + (size_t)B * H * HD * HD;     // (B*H, 16)
  _Float16* wq_h = (_Float16*)(ksum + (size_t)B * H * HD);
  _Float16* wp_h = wq_h + (size_t)C3 * C;

  cvt_weights_kernel<<<(C3 * C + 255) / 256, 256, 0, stream>>>(w_qkv, w_proj,
                                                               wq_h, wp_h);
  qkv_focus_kernel<<<BN / TM, 256, 0, stream>>>(x, pos, mask, scale_param,
                                                wq_h, b_qkv, qh, kh, vh);
  kv_ksum_kernel<<<B * H, 256, 0, stream>>>(kh, vh, kv_g, ksum);
  attn_out_kernel<<<B * H, 256, 0, stream>>>(qh, kv_g, ksum, vox, o_pre);
  proj_kernel<<<BN / TM, 256, 0, stream>>>(o_pre, wp_h, b_proj, out);
}
